// MultiHeadAttention_54580444397743
// MI455X (gfx1250) — matrix operand order repaired and checked by reading, not compiled, not run on hardware
//
#include <hip/hip_runtime.h>

// ---------------------------------------------------------------------------
// Multi-head attention for MI455X (gfx1250).  bf16 WMMA (v_wmma_f32_16x16x32_bf16),
// f32 accumulate, f32 softmax.  B=2, T=2048, C=1024, H=16, D=64.
// Operands pre-packed to bf16 with K-contiguous per-lane layouts: every WMMA
// fragment = two 16-byte loads.  GEMMs use 32x64 tiles with a software-
// pipelined K-loop; attention uses 32-query tiles with hoisted V loads and
// prefetch of the next key block.
// One wave32 per block => EXEC all ones around every WMMA (ISA requirement).
// ---------------------------------------------------------------------------

typedef __bf16 bf16_t;
typedef __attribute__((ext_vector_type(16))) __bf16 v16bf;
typedef __attribute__((ext_vector_type(8)))  __bf16 v8bf;
typedef __attribute__((ext_vector_type(8)))  float  v8f;

#define B_  2
#define T_  2048
#define C_  1024
#define H_  16
#define D_  64
#define BT_ (B_ * T_)
#define SCALE_ 32.0f   // reference multiplies scores by sqrt(C) = sqrt(1024)

// A-matrix (16x32 bf16): lane holds row M=lane&15; element j -> K:
//   j in [0,8):  K = half*8 + j            j in [8,16): K = 16 + half*8 + (j-8)
// B-matrix (32x16 bf16): lane holds col N=lane&15; element j -> K same as A
// C/D (16x16 f32): VGPR r -> M = r + half*8, N = lane&15.

static __device__ __forceinline__ v16bf join16(v8bf lo, v8bf hi) {
  return __builtin_shufflevector(lo, hi, 0, 1, 2, 3, 4, 5, 6, 7,
                                         8, 9, 10, 11, 12, 13, 14, 15);
}
static __device__ __forceinline__ v16bf load_a(const bf16_t* row, int kb, int half) {
  v8bf lo = *(const v8bf*)(row + kb + (half << 3));
  v8bf hi = *(const v8bf*)(row + kb + 16 + (half << 3));
  return join16(lo, hi);
}
static __device__ __forceinline__ v16bf load_b(const bf16_t* col, int kb, int half) {
  v8bf lo = *(const v8bf*)(col + kb + (half << 3));
  v8bf hi = *(const v8bf*)(col + kb + 16 + (half << 3));
  return join16(lo, hi);
}
static __device__ __forceinline__ v8f wmma_bf16(v16bf a, v16bf b, v8f c) {
  return __builtin_amdgcn_wmma_f32_16x16x32_bf16(
      false, a, false, b, (short)0, c, false, false);
}

// ---------------------------------------------------------------------------
// Prep 1: ix f32 -> bf16 (coalesced).
// ---------------------------------------------------------------------------
__global__ __launch_bounds__(256) void cvt_ix_kernel(
    const float* __restrict__ ix, bf16_t* __restrict__ ixbf)
{
  const size_t i = (size_t)blockIdx.x * 256 + threadIdx.x;
  ixbf[i] = (bf16_t)ix[i];
}

// ---------------------------------------------------------------------------
// Prep 2: pack weights to bf16, transposed so output-column is K-contiguous.
//   wqkvT[m][col][c] = W_m[h = col>>6][c][d = col&63]     (m = 0,1,2)
//   wpT[col][c]      = Wp[c][col]
// ---------------------------------------------------------------------------
__global__ __launch_bounds__(256) void pack_w_kernel(
    const float* __restrict__ Wq, const float* __restrict__ Wk,
    const float* __restrict__ Wv, const float* __restrict__ Wp,
    bf16_t* __restrict__ wqkvT, bf16_t* __restrict__ wpT)
{
  const int idx = blockIdx.x * 256 + threadIdx.x;  // 0 .. 1M-1
  const int col = idx >> 10;
  const int c   = idx & (C_ - 1);
  const int m   = blockIdx.y;                      // 0=Q 1=K 2=V 3=P
  if (m < 3) {
    const float* W = (m == 0) ? Wq : (m == 1) ? Wk : Wv;
    const int h = col >> 6, d = col & 63;
    wqkvT[(size_t)m * C_ * C_ + idx] = (bf16_t)W[((size_t)h * C_ + c) * D_ + d];
  } else {
    wpT[idx] = (bf16_t)Wp[(size_t)c * C_ + col];
  }
}

// ---------------------------------------------------------------------------
// Kernel 1: fused QKV projection.  32x64 output tile per wave; K-loop
// software-pipelined (next fragments preloaded before current WMMAs).
// Q,K stored [B,H,T,D]; V stored transposed [B,H,D,T].
// ---------------------------------------------------------------------------
__global__ __launch_bounds__(32) void qkv_proj_kernel(
    const bf16_t* __restrict__ ixbf, const bf16_t* __restrict__ wqkvT,
    bf16_t* __restrict__ qbf, bf16_t* __restrict__ kbf, bf16_t* __restrict__ vT)
{
  const int lane = threadIdx.x;
  const int half = lane >> 4, l16 = lane & 15;
  const int rowBase = blockIdx.x << 5;          // 128 tiles over 4096 rows
  const int cTile   = blockIdx.y;               // 0..47
  const int which   = cTile >> 4;               // 0=Q 1=K 2=V
  const int wcol    = (cTile & 15) << 6;
  const int h       = wcol >> 6;

  const bf16_t* arow0 = ixbf + (size_t)(rowBase + l16) * C_;
  const bf16_t* arow1 = ixbf + (size_t)(rowBase + 16 + l16) * C_;
  const bf16_t* wmat  = wqkvT + (size_t)which * C_ * C_;
  const bf16_t* bcol[4];
#pragma unroll
  for (int n = 0; n < 4; ++n)
    bcol[n] = wmat + (size_t)(wcol + n * 16 + l16) * C_;

  v8f acc[2][4] = {{{}, {}, {}, {}}, {{}, {}, {}, {}}};
  v16bf a0 = load_a(arow0, 0, half), a1 = load_a(arow1, 0, half);
  v16bf b0 = load_b(bcol[0], 0, half), b1 = load_b(bcol[1], 0, half);
  v16bf b2 = load_b(bcol[2], 0, half), b3 = load_b(bcol[3], 0, half);

  for (int kb = 0; kb < C_ - 32; kb += 32) {
    const int nk = kb + 32;
    v16bf na0 = load_a(arow0, nk, half), na1 = load_a(arow1, nk, half);
    v16bf nb0 = load_b(bcol[0], nk, half), nb1 = load_b(bcol[1], nk, half);
    v16bf nb2 = load_b(bcol[2], nk, half), nb3 = load_b(bcol[3], nk, half);
    acc[0][0] = wmma_bf16(a0, b0, acc[0][0]);
    acc[1][0] = wmma_bf16(a1, b0, acc[1][0]);
    acc[0][1] = wmma_bf16(a0, b1, acc[0][1]);
    acc[1][1] = wmma_bf16(a1, b1, acc[1][1]);
    acc[0][2] = wmma_bf16(a0, b2, acc[0][2]);
    acc[1][2] = wmma_bf16(a1, b2, acc[1][2]);
    acc[0][3] = wmma_bf16(a0, b3, acc[0][3]);
    acc[1][3] = wmma_bf16(a1, b3, acc[1][3]);
    a0 = na0; a1 = na1; b0 = nb0; b1 = nb1; b2 = nb2; b3 = nb3;
  }
  acc[0][0] = wmma_bf16(a0, b0, acc[0][0]);
  acc[1][0] = wmma_bf16(a1, b0, acc[1][0]);
  acc[0][1] = wmma_bf16(a0, b1, acc[0][1]);
  acc[1][1] = wmma_bf16(a1, b1, acc[1][1]);
  acc[0][2] = wmma_bf16(a0, b2, acc[0][2]);
  acc[1][2] = wmma_bf16(a1, b2, acc[1][2]);
  acc[0][3] = wmma_bf16(a0, b3, acc[0][3]);
  acc[1][3] = wmma_bf16(a1, b3, acc[1][3]);

#pragma unroll
  for (int sub = 0; sub < 2; ++sub)
#pragma unroll
    for (int r = 0; r < 8; ++r) {
      const int row  = rowBase + sub * 16 + r + (half << 3);
      const int bidx = row >> 11;
      const int t    = row & (T_ - 1);
      const size_t bh = (size_t)bidx * H_ + h;
#pragma unroll
      for (int n = 0; n < 4; ++n) {
        const int d = n * 16 + l16;
        const bf16_t v = (bf16_t)acc[sub][n][r];
        if (which == 0)      qbf[(bh * T_ + t) * D_ + d] = v;
        else if (which == 1) kbf[(bh * T_ + t) * D_ + d] = v;
        else                 vT[(bh * D_ + d) * T_ + t]  = v;
      }
    }
}

// ---------------------------------------------------------------------------
// Kernel 2: causal attention, online softmax.  One wave = 32 query rows of
// one (b,h); 32 keys per iteration: 8 WMMAs for S = Q@K^T, 8 for P@V.
// V loads hoisted above the score WMMAs; next key block prefetched.
// ---------------------------------------------------------------------------
__global__ __launch_bounds__(32) void attn_kernel(
    const bf16_t* __restrict__ qbf, const bf16_t* __restrict__ kbf,
    const bf16_t* __restrict__ vT, bf16_t* __restrict__ att)
{
  __shared__ bf16_t ldsP[32 * 32];             // P tile, row-major 32x32
  const int lane = threadIdx.x;
  const int half = lane >> 4, l16 = lane & 15;
  const int qT = blockIdx.x;                   // 0..63
  const int bh = blockIdx.y;                   // 0..31
  const int b  = bh >> 4, h = bh & 15;
  const size_t base = (size_t)bh * T_;
  const int qBase = qT << 5;

  // Q fragments: [qsub][kc]
  v16bf qf[2][2];
#pragma unroll
  for (int sub = 0; sub < 2; ++sub) {
    const bf16_t* qrow = qbf + (base + qBase + sub * 16 + l16) * D_;
    qf[sub][0] = load_a(qrow, 0, half);
    qf[sub][1] = load_a(qrow, 32, half);
  }

  v8f o[2][4] = {{{}, {}, {}, {}}, {{}, {}, {}, {}}};
  float mrow[2][8], lrow[2][8];
#pragma unroll
  for (int sub = 0; sub < 2; ++sub)
#pragma unroll
    for (int r = 0; r < 8; ++r) { mrow[sub][r] = -__builtin_inff(); lrow[sub][r] = 0.f; }

  for (int sBase = 0; sBase <= qBase + 31; sBase += 32) {
    const bf16_t* krow0 = kbf + (base + sBase +      l16) * D_;
    const bf16_t* krow1 = kbf + (base + sBase + 16 + l16) * D_;

    // prefetch next key block (K rows + both halves of V^T columns)
    __builtin_prefetch(krow0 + 32 * D_, 0, 0);
    __builtin_prefetch(vT + ((size_t)bh * D_ + lane) * T_ + sBase + 32, 0, 0);
    __builtin_prefetch(vT + ((size_t)bh * D_ + 32 + lane) * T_ + sBase + 32, 0, 0);

    // hoist V fragments: in flight during score WMMAs + softmax VALU work
    v16bf vb[4];
#pragma unroll
    for (int dT = 0; dT < 4; ++dT) {
      const bf16_t* vcol = vT + ((size_t)bh * D_ + dT * 16 + l16) * T_ + sBase;
      v8bf lo = *(const v8bf*)(vcol + (half << 3));
      v8bf hi = *(const v8bf*)(vcol + 16 + (half << 3));
      vb[dT] = join16(lo, hi);
    }

    // ---- S = Q @ K^T : [qsub][ksub] score tiles ----
    v8f s[2][2] = {{{}, {}}, {{}, {}}};
#pragma unroll
    for (int kc = 0; kc < 2; ++kc) {
      const v16bf kf0 = load_b(krow0, kc * 32, half);
      const v16bf kf1 = load_b(krow1, kc * 32, half);
#pragma unroll
      for (int sub = 0; sub < 2; ++sub) {
        s[sub][0] = wmma_bf16(qf[sub][kc], kf0, s[sub][0]);
        s[sub][1] = wmma_bf16(qf[sub][kc], kf1, s[sub][1]);
      }
    }

    // ---- scale by sqrt(C), causal mask, online softmax, stage P in LDS ----
    const int kj0 = sBase + l16, kj1 = sBase + 16 + l16;
#pragma unroll
    for (int sub = 0; sub < 2; ++sub) {
      float p0[8], p1[8], mnew[8];
#pragma unroll
      for (int r = 0; r < 8; ++r) {
        const int qi = qBase + sub * 16 + r + (half << 3);
        float a0 = (kj0 <= qi) ? s[sub][0][r] * SCALE_ : -__builtin_inff();
        float a1 = (kj1 <= qi) ? s[sub][1][r] * SCALE_ : -__builtin_inff();
        p0[r] = a0; p1[r] = a1;
        float m = fmaxf(a0, a1);
#pragma unroll
        for (int x = 1; x < 16; x <<= 1)           // reduce over 16 N-lanes
          m = fmaxf(m, __shfl_xor(m, x, 32));
        mnew[r] = fmaxf(mrow[sub][r], m);
      }
#pragma unroll
      for (int r = 0; r < 8; ++r) {
        float e0 = __expf(p0[r] - mnew[r]);
        float e1 = __expf(p1[r] - mnew[r]);
        float srow = e0 + e1;
#pragma unroll
        for (int x = 1; x < 16; x <<= 1)
          srow += __shfl_xor(srow, x, 32);
        const float f = __expf(mrow[sub][r] - mnew[sub == 0 ? r : r] * 1.0f);
        // (f uses old mrow vs new max)
        const float fac = __expf(mrow[sub][r] - mnew[r]);
        lrow[sub][r] = lrow[sub][r] * fac + srow;
        mrow[sub][r] = mnew[r];
        o[sub][0][r] *= fac; o[sub][1][r] *= fac;
        o[sub][2][r] *= fac; o[sub][3][r] *= fac;
        const int lrowIdx = (sub * 16 + r + (half << 3)) * 32;
        ldsP[lrowIdx + l16]      = (bf16_t)e0;
        ldsP[lrowIdx + l16 + 16] = (bf16_t)e1;
        (void)f;
      }
    }
    __syncthreads();
    v16bf pA[2];
    pA[0] = load_a(ldsP + l16 * 32, 0, half);
    pA[1] = load_a(ldsP + (16 + l16) * 32, 0, half);
    __syncthreads();

    // ---- O += P @ V ----
#pragma unroll
    for (int dT = 0; dT < 4; ++dT)
#pragma unroll
      for (int sub = 0; sub < 2; ++sub)
        o[sub][dT] = wmma_bf16(pA[sub], vb[dT], o[sub][dT]);
  }

  // ---- normalize, store concat output att[b, t, h*64 + d] ----
#pragma unroll
  for (int sub = 0; sub < 2; ++sub)
#pragma unroll
    for (int dT = 0; dT < 4; ++dT)
#pragma unroll
      for (int r = 0; r < 8; ++r) {
        const int t = qBase + sub * 16 + r + (half << 3);
        att[((size_t)b * T_ + t) * C_ + h * D_ + dT * 16 + l16] =
            (bf16_t)(o[sub][dT][r] / lrow[sub][r]);
      }
}

// ---------------------------------------------------------------------------
// Kernel 3: Y = att @ Wp + bp (f32 out).  32x64 tile per wave, pipelined.
// ---------------------------------------------------------------------------
__global__ __launch_bounds__(32) void out_proj_kernel(
    const bf16_t* __restrict__ att, const bf16_t* __restrict__ wpT,
    const float* __restrict__ bp, float* __restrict__ y)
{
  const int lane = threadIdx.x;
  const int half = lane >> 4, l16 = lane & 15;
  const int rowBase = blockIdx.x << 5;          // 128 tiles
  const int colBase = blockIdx.y << 6;          // 16 tiles of 64

  const bf16_t* arow0 = att + (size_t)(rowBase + l16) * C_;
  const bf16_t* arow1 = att + (size_t)(rowBase + 16 + l16) * C_;
  const bf16_t* bcol[4];
#pragma unroll
  for (int n = 0; n < 4; ++n)
    bcol[n] = wpT + (size_t)(colBase + n * 16 + l16) * C_;

  v8f acc[2][4] = {{{}, {}, {}, {}}, {{}, {}, {}, {}}};
  v16bf a0 = load_a(arow0, 0, half), a1 = load_a(arow1, 0, half);
  v16bf b0 = load_b(bcol[0], 0, half), b1 = load_b(bcol[1], 0, half);
  v16bf b2 = load_b(bcol[2], 0, half), b3 = load_b(bcol[3], 0, half);

  for (int kb = 0; kb < C_ - 32; kb += 32) {
    const int nk = kb + 32;
    v16bf na0 = load_a(arow0, nk, half), na1 = load_a(arow1, nk, half);
    v16bf nb0 = load_b(bcol[0], nk, half), nb1 = load_b(bcol[1], nk, half);
    v16bf nb2 = load_b(bcol[2], nk, half), nb3 = load_b(bcol[3], nk, half);
    acc[0][0] = wmma_bf16(a0, b0, acc[0][0]);
    acc[1][0] = wmma_bf16(a1, b0, acc[1][0]);
    acc[0][1] = wmma_bf16(a0, b1, acc[0][1]);
    acc[1][1] = wmma_bf16(a1, b1, acc[1][1]);
    acc[0][2] = wmma_bf16(a0, b2, acc[0][2]);
    acc[1][2] = wmma_bf16(a1, b2, acc[1][2]);
    acc[0][3] = wmma_bf16(a0, b3, acc[0][3]);
    acc[1][3] = wmma_bf16(a1, b3, acc[1][3]);
    a0 = na0; a1 = na1; b0 = nb0; b1 = nb1; b2 = nb2; b3 = nb3;
  }
  acc[0][0] = wmma_bf16(a0, b0, acc[0][0]);
  acc[1][0] = wmma_bf16(a1, b0, acc[1][0]);
  acc[0][1] = wmma_bf16(a0, b1, acc[0][1]);
  acc[1][1] = wmma_bf16(a1, b1, acc[1][1]);
  acc[0][2] = wmma_bf16(a0, b2, acc[0][2]);
  acc[1][2] = wmma_bf16(a1, b2, acc[1][2]);
  acc[0][3] = wmma_bf16(a0, b3, acc[0][3]);
  acc[1][3] = wmma_bf16(a1, b3, acc[1][3]);

#pragma unroll
  for (int n = 0; n < 4; ++n) {
    const int col = colBase + n * 16 + l16;
    const float bias = bp[col];
#pragma unroll
    for (int sub = 0; sub < 2; ++sub)
#pragma unroll
      for (int r = 0; r < 8; ++r) {
        const int row = rowBase + sub * 16 + r + (half << 3);
        y[(size_t)row * C_ + col] = acc[sub][n][r] + bias;
      }
  }
}

// ---------------------------------------------------------------------------
extern "C" void kernel_launch(void* const* d_in, const int* in_sizes, int n_in,
                              void* d_out, int out_size, void* d_ws, size_t ws_size,
                              hipStream_t stream) {
  const float* ix = (const float*)d_in[0];
  const float* Wq = (const float*)d_in[1];
  const float* Wk = (const float*)d_in[2];
  const float* Wv = (const float*)d_in[3];
  const float* Wp = (const float*)d_in[4];
  const float* bp = (const float*)d_in[5];
  float* y = (float*)d_out;

  char* ws = (char*)d_ws;
  const size_t MiB = 1024 * 1024;
  bf16_t* ixbf  = (bf16_t*)(ws);                 //  8 MiB  [BT, C]
  bf16_t* wqkvT = (bf16_t*)(ws + 8 * MiB);       //  6 MiB  [3, 1024, 1024]
  bf16_t* wpT   = (bf16_t*)(ws + 14 * MiB);      //  2 MiB  [1024, 1024]
  bf16_t* qbf   = (bf16_t*)(ws + 16 * MiB);      //  8 MiB  [B,H,T,D]
  bf16_t* kbf   = (bf16_t*)(ws + 24 * MiB);      //  8 MiB  [B,H,T,D]
  bf16_t* vT    = (bf16_t*)(ws + 32 * MiB);      //  8 MiB  [B,H,D,T]
  bf16_t* att   = (bf16_t*)(ws + 40 * MiB);      //  8 MiB  [B,T,C]

  cvt_ix_kernel<<<(BT_ * C_) / 256, 256, 0, stream>>>(ix, ixbf);
  pack_w_kernel<<<dim3((C_ * C_) / 256, 4), 256, 0, stream>>>(
      Wq, Wk, Wv, Wp, wqkvT, wpT);
  qkv_proj_kernel<<<dim3(BT_ / 32, 48), 32, 0, stream>>>(
      ixbf, wqkvT, qbf, kbf, vT);
  attn_kernel<<<dim3(T_ / 32, B_ * H_), 32, 0, stream>>>(qbf, kbf, vT, att);
  out_proj_kernel<<<dim3(BT_ / 32, 16), 32, 0, stream>>>(att, wpT, bp, y);
}
